// NSF_89910845374663
// MI455X (gfx1250) — compile-verified, matrix operand order checked by reading, not run on hardware
//
#include <hip/hip_runtime.h>
#include <hip/hip_bf16.h>
#include <math.h>

// ---------------- problem constants ----------------
#define NROWS   8192        // 16*512
#define LATD    16
#define CTXD    256
#define HIDD    256
#define NBLK    8
#define NBINS   8
#define D_ID    8
#define D_TR    8
#define POUT    184         // D_TR*(3*NBINS-1)
#define TAILF   3.0f
#define MINWH   0.001f
#define KIN     264         // D_ID + CTX
#define KINPAD  288         // padded to multiple of 32

typedef __attribute__((ext_vector_type(4)))  _Float16 v4h;
typedef __attribute__((ext_vector_type(8)))  _Float16 v8h;
typedef __attribute__((ext_vector_type(16))) _Float16 v16h;
typedef __attribute__((ext_vector_type(8)))  float    v8f;

// ---------------- spline helpers ----------------
struct Spline {
    float cw[9], ww[8], ch[9], hh[8], dd[9];
};

__device__ inline float softplusf(float x) {
    return (x > 20.f) ? x : log1pf(expf(x));
}

__device__ inline void build_spline(const float* uw, const float* uh,
                                    const float* ud, float scale, Spline* s) {
    // widths
    {
        float m = -1e30f;
        for (int k = 0; k < 8; ++k) m = fmaxf(m, uw[k] * scale);
        float e[8], sum = 0.f;
        for (int k = 0; k < 8; ++k) { e[k] = expf(uw[k] * scale - m); sum += e[k]; }
        float c = 0.f;
        s->cw[0] = -TAILF;
        for (int k = 0; k < 8; ++k) {
            float w = MINWH + (1.f - 8.f * MINWH) * (e[k] / sum);
            c += w;
            s->cw[k + 1] = 2.f * TAILF * c - TAILF;
        }
        s->cw[8] = TAILF;
        for (int k = 0; k < 8; ++k) s->ww[k] = s->cw[k + 1] - s->cw[k];
    }
    // heights
    {
        float m = -1e30f;
        for (int k = 0; k < 8; ++k) m = fmaxf(m, uh[k] * scale);
        float e[8], sum = 0.f;
        for (int k = 0; k < 8; ++k) { e[k] = expf(uh[k] * scale - m); sum += e[k]; }
        float c = 0.f;
        s->ch[0] = -TAILF;
        for (int k = 0; k < 8; ++k) {
            float h = MINWH + (1.f - 8.f * MINWH) * (e[k] / sum);
            c += h;
            s->ch[k + 1] = 2.f * TAILF * c - TAILF;
        }
        s->ch[8] = TAILF;
        for (int k = 0; k < 8; ++k) s->hh[k] = s->ch[k + 1] - s->ch[k];
    }
    // padded end derivatives: softplus(DCONST) = 1-MIND -> d = 1.0 exactly
    s->dd[0] = 1.f;
    s->dd[8] = 1.f;
    for (int k = 0; k < 7; ++k) s->dd[k + 1] = MINWH + softplusf(ud[k]);
}

__device__ inline float rqs_inv(float x, const Spline& s, float* dld) {
    bool inside = (x >= -TAILF) && (x <= TAILF);
    float xc = fminf(fmaxf(x, -TAILF), TAILF);
    int idx = 0;
    for (int k = 0; k < 9; ++k) {
        float edge = s.ch[k] + ((k == 8) ? 1e-6f : 0.f);
        idx += (xc >= edge) ? 1 : 0;
    }
    idx -= 1;
    idx = min(max(idx, 0), 7);
    float cwv = s.cw[idx], wv = s.ww[idx];
    float chv = s.ch[idx], hv = s.hh[idx];
    float delta = hv / wv;
    float d0 = s.dd[idx], d1 = s.dd[idx + 1];
    float sv = d0 + d1 - 2.f * delta;
    float y  = xc - chv;
    float a  = y * sv + hv * (delta - d0);
    float b  = hv * d0 - y * sv;
    float cc = -delta * y;
    float disc = fmaxf(b * b - 4.f * a * cc, 0.f);
    float root = 2.f * cc / (-b - sqrtf(disc));
    float outv = root * wv + cwv;
    float t1m = root * (1.f - root);
    float den = delta + sv * t1m;
    float num = delta * delta *
                (d1 * root * root + 2.f * delta * t1m + d0 * (1.f - root) * (1.f - root));
    float lad = logf(num) - 2.f * logf(den);
    *dld = inside ? -lad : 0.f;
    return inside ? outv : x;
}

// ---------------- elementwise kernels ----------------
__global__ void init_state(const float* __restrict__ x,
                           float* __restrict__ zbuf, float* __restrict__ logdet) {
    int row = blockIdx.x * blockDim.x + threadIdx.x;
    if (row >= NROWS) return;
    for (int d = 0; d < LATD; ++d) zbuf[row * LATD + d] = x[row * LATD + d];
    logdet[row] = 0.f;
}

__global__ void build_uncond(const float* __restrict__ uw, const float* __restrict__ uh,
                             const float* __restrict__ ud, Spline* __restrict__ uspl) {
    int t = threadIdx.x;                     // blk*8 + dim
    if (t >= NBLK * D_ID) return;
    Spline s;
    build_spline(uw + t * 8, uh + t * 8, ud + t * 7, 1.f, &s);
    uspl[t] = s;
}

// one-time weight convert: src f32 [K][Msrc] -> dst f16 [Mpad][Kpad] (k-contiguous)
__global__ void conv_w16(const float* __restrict__ src, _Float16* __restrict__ dst,
                         int K, int Msrc, int Kpad) {
    int idx = blockIdx.x * 256 + threadIdx.x;    // over Mpad*Kpad
    int col = idx / Kpad, k = idx % Kpad;
    float v = (col < Msrc && k < K) ? src[k * Msrc + col] : 0.f;
    dst[idx] = (_Float16)v;
}

__global__ void prep_block(const float* __restrict__ zbuf,
                           const int* __restrict__ perm,
                           const float* __restrict__ lu_lo,
                           const float* __restrict__ lu_up,
                           const float* __restrict__ lu_ud,
                           const float* __restrict__ lu_b,
                           const Spline* __restrict__ uspl,
                           float* __restrict__ zidb, float* __restrict__ ztrb,
                           float* __restrict__ logdet) {
    int row = blockIdx.x * blockDim.x + threadIdx.x;
    if (row >= NROWS) return;
    const float* zr = zbuf + row * LATD;
    float zp[LATD];
    for (int d = 0; d < LATD; ++d) zp[d] = zr[perm[d]];

    float diag[LATD], ldc = 0.f;
    for (int r = 0; r < LATD; ++r) {
        diag[r] = softplusf(lu_ud[r]) + 0.001f;
        ldc += logf(diag[r]);
    }
    float y[LATD];
    int off = 0;
    for (int r = 0; r < LATD; ++r) {          // y = U * zp  (triu row-major)
        float acc = diag[r] * zp[r];
        for (int c = r + 1; c < LATD; ++c) acc += lu_up[off++] * zp[c];
        y[r] = acc;
    }
    float zn[LATD];
    off = 0;
    for (int r = 0; r < LATD; ++r) {          // z = L * y + bias  (tril row-major)
        float acc = y[r];
        for (int c = 0; c < r; ++c) acc += lu_lo[off++] * y[c];
        zn[r] = acc + lu_b[r];
    }
    float lad = 0.f;
    for (int d = 0; d < D_ID; ++d) {
        float dl;
        float v = rqs_inv(zn[2 * d], uspl[d], &dl);
        zidb[row * D_ID + d] = v;
        ztrb[row * D_TR + d] = zn[2 * d + 1];
        lad += dl;
    }
    logdet[row] += ldc + lad;
}

__global__ void finish_block(const float* __restrict__ pbuf,
                             const float* __restrict__ zidb,
                             const float* __restrict__ ztrb,
                             float* __restrict__ zbuf, float* __restrict__ logdet) {
    int row = blockIdx.x * blockDim.x + threadIdx.x;
    if (row >= NROWS) return;
    const float* pr = pbuf + row * POUT;
    const float invSqrtH = 0.0625f;          // 1/sqrt(256)
    float lad = 0.f;
    for (int d = 0; d < D_TR; ++d) {
        Spline s;
        build_spline(pr + d * 23, pr + d * 23 + 8, pr + d * 23 + 16, invSqrtH, &s);
        float dl;
        float t = rqs_inv(ztrb[row * D_TR + d], s, &dl);
        zbuf[row * LATD + 2 * d]     = zidb[row * D_ID + d];
        zbuf[row * LATD + 2 * d + 1] = t;
        lad += dl;
    }
    logdet[row] += lad;
}

__global__ void final_gauss(const float* __restrict__ ctx,
                            const float* __restrict__ q0W,
                            const float* __restrict__ q0b,
                            const float* __restrict__ zbuf,
                            const float* __restrict__ logdet,
                            float* __restrict__ out) {
    int row = blockIdx.x * blockDim.x + threadIdx.x;
    if (row >= NROWS) return;
    const float* c = ctx + row * CTXD;
    float enc[2 * LATD];
    for (int o = 0; o < 2 * LATD; ++o) enc[o] = q0b[o];
    for (int k = 0; k < CTXD; ++k) {
        float cv = c[k];
        const float* wr = q0W + k * (2 * LATD);
        for (int o = 0; o < 2 * LATD; ++o) enc[o] += cv * wr[o];
    }
    float logp = -14.703016548f;             // -0.5 * 16 * log(2*pi)
    for (int d = 0; d < LATD; ++d) {
        float ls = enc[LATD + d];
        float zz = (zbuf[row * LATD + d] - enc[d]) * expf(-ls);
        logp -= ls + 0.5f * zz * zz;
    }
    out[row] = -(logp + logdet[row]);
}

// ---------------- WMMA GEMM ----------------
// Tile: 64 rows x 256 cols per workgroup (8 waves; each wave 32x64 = 8 accumulators).
// A staged f32->f16 via float4 loads; W pre-converted f16 [Mpad=256][Kpad], staged
// with global_load_async_to_lds_b128 (ASYNCcnt, gfx1250 async path).
// MODE 0: Out = acc + bias ; MODE 1: Out += (acc+bias)*sigmoid(gate)
#define LDS_STRIDE 40   // halves; 80 B row pitch -> all fragment loads 16 B aligned

__device__ inline unsigned lds_addr32(const _Float16* p) {
    return (unsigned)(unsigned long long)
           (__attribute__((address_space(3))) const _Float16*)p;
}

__device__ inline v16h frag_ld(const _Float16* rowp, int halfSel) {
    v8h lo = *(const v8h*)(rowp + halfSel);
    v8h hi = *(const v8h*)(rowp + 16 + halfSel);
    return __builtin_shufflevector(lo, hi, 0, 1, 2, 3, 4, 5, 6, 7,
                                   8, 9, 10, 11, 12, 13, 14, 15);
}

template <int MODE, bool RELUA, bool CONCAT>
__launch_bounds__(256)
__global__ void gemm_wmma(const float* __restrict__ A, const float* __restrict__ A2,
                          const _Float16* __restrict__ Wt,  // [256][Kpad] f16
                          const float* __restrict__ bias,
                          const float* __restrict__ gate, float* __restrict__ out,
                          int K, int Kpad, int M) {
    __shared__ __align__(16) _Float16 As[64 * LDS_STRIDE];
    __shared__ __align__(16) _Float16 Ws[256 * LDS_STRIDE];

    const int tid  = threadIdx.x;
    const int lane = tid & 31;
    const int wave = tid >> 5;
    const int rowBase = blockIdx.y * 64;
    const int waveRow = (wave >> 2) * 32;       // 0 / 32
    const int waveCol = (wave & 3) * 64;        // 0..192
    const int halfSel = (lane >= 16) ? 8 : 0;
    const int mRow    = lane & 15;

    const unsigned long long wbase = (unsigned long long)Wt;

    v8f acc0[4] = {};
    v8f acc1[4] = {};

    const int ksteps = Kpad >> 5;
    for (int ks = 0; ks < ksteps; ++ks) {
        const int kbase = ks * 32;

        // ---- stage A: 64 rows x 32 k = 512 float4 chunks (2 per thread) ----
#pragma unroll
        for (int it = 0; it < 2; ++it) {
            int idx = it * 256 + tid;
            int rr  = idx >> 3;               // 0..63
            int k4  = idx & 7;                // chunk-of-4 within k tile
            int kg  = kbase + k4 * 4;
            float4 v = make_float4(0.f, 0.f, 0.f, 0.f);
            if (CONCAT) {
                if (kg < 8)
                    v = *(const float4*)(A + (rowBase + rr) * 8 + kg);
                else if (kg < K)
                    v = *(const float4*)(A2 + (rowBase + rr) * 256 + (kg - 8));
            } else {
                v = *(const float4*)(A + (rowBase + rr) * K + kg);
            }
            if (RELUA) {
                v.x = fmaxf(v.x, 0.f); v.y = fmaxf(v.y, 0.f);
                v.z = fmaxf(v.z, 0.f); v.w = fmaxf(v.w, 0.f);
            }
            v4h h = { (_Float16)v.x, (_Float16)v.y, (_Float16)v.z, (_Float16)v.w };
            *(v4h*)&As[rr * LDS_STRIDE + k4 * 4] = h;
        }

        // ---- stage W: async DMA, 256 cols x 32 halves = 1024 x 16 B chunks ----
#pragma unroll
        for (int it = 0; it < 4; ++it) {
            int idx = it * 256 + tid;
            int col = idx >> 2;               // 0..255
            int kc  = idx & 3;                // 8-half chunk within k tile
            unsigned ldsa = lds_addr32(&Ws[col * LDS_STRIDE + kc * 8]);
            unsigned goff = (unsigned)((col * Kpad + kbase + kc * 8) * 2);
            asm volatile("global_load_async_to_lds_b128 %0, %1, %2"
                         :: "v"(ldsa), "v"(goff), "s"(wbase) : "memory");
        }
        // prefetch next A tile (speculative global_prefetch_b8)
        if (!CONCAT && ks + 1 < ksteps)
            __builtin_prefetch(A + (rowBase + (tid >> 2)) * K + kbase + 32, 0, 1);

        asm volatile("s_wait_asynccnt 0x0" ::: "memory");
        __syncthreads();

        // ---- fragments + WMMA (B fragment reused across both row tiles) ----
        v16h fa0 = frag_ld(&As[(waveRow + mRow) * LDS_STRIDE], halfSel);
        v16h fa1 = frag_ld(&As[(waveRow + 16 + mRow) * LDS_STRIDE], halfSel);
#pragma unroll
        for (int f = 0; f < 4; ++f) {
            v16h fb = frag_ld(&Ws[(waveCol + f * 16 + mRow) * LDS_STRIDE], halfSel);
            acc0[f] = __builtin_amdgcn_wmma_f32_16x16x32_f16(
                          false, fa0, false, fb, (short)0, acc0[f], false, false);
            acc1[f] = __builtin_amdgcn_wmma_f32_16x16x32_f16(
                          false, fa1, false, fb, (short)0, acc1[f], false, false);
        }
        __syncthreads();
    }

    // ---- epilogue ----
#pragma unroll
    for (int rt = 0; rt < 2; ++rt) {
        const v8f* ac = rt ? acc1 : acc0;
#pragma unroll
        for (int r = 0; r < 8; ++r) {
            int row = rowBase + waveRow + rt * 16 + ((lane >= 16) ? 8 : 0) + r;
#pragma unroll
            for (int f = 0; f < 4; ++f) {
                int col = waveCol + f * 16 + mRow;
                if (col >= M) continue;
                int idx = row * M + col;
                float v = ac[f][r] + bias[col];
                if (MODE == 1) {
                    float sg = 1.f / (1.f + expf(-gate[idx]));
                    out[idx] = out[idx] + v * sg;
                } else {
                    out[idx] = v;
                }
            }
        }
    }
}

// ---------------- host orchestration ----------------
extern "C" void kernel_launch(void* const* d_in, const int* in_sizes, int n_in,
                              void* d_out, int out_size, void* d_ws, size_t ws_size,
                              hipStream_t stream) {
    const float* x     = (const float*)d_in[0];
    const float* ctx   = (const float*)d_in[1];
    const float* W_in  = (const float*)d_in[2];
    const float* b_in  = (const float*)d_in[3];
    const float* W1    = (const float*)d_in[4];
    const float* b1    = (const float*)d_in[5];
    const float* W2    = (const float*)d_in[6];
    const float* b2    = (const float*)d_in[7];
    const float* Wc    = (const float*)d_in[8];
    const float* bc    = (const float*)d_in[9];
    const float* Wf    = (const float*)d_in[10];
    const float* bf    = (const float*)d_in[11];
    const float* unc_w = (const float*)d_in[12];
    const float* unc_h = (const float*)d_in[13];
    const float* unc_d = (const float*)d_in[14];
    const float* lu_lo = (const float*)d_in[15];
    const float* lu_up = (const float*)d_in[16];
    const float* lu_ud = (const float*)d_in[17];
    const float* lu_b  = (const float*)d_in[18];
    const float* q0W   = (const float*)d_in[19];
    const float* q0b   = (const float*)d_in[20];
    const int*   perms = (const int*)d_in[21];
    float* out = (float*)d_out;

    // workspace carve-up
    float* ws = (float*)d_ws;
    size_t off = 0;
    float* zbuf   = ws + off; off += (size_t)NROWS * LATD;
    float* zidb   = ws + off; off += (size_t)NROWS * D_ID;
    float* ztrb   = ws + off; off += (size_t)NROWS * D_TR;
    float* logdet = ws + off; off += (size_t)NROWS;
    float* hcur   = ws + off; off += (size_t)NROWS * HIDD;
    float* t1b    = ws + off; off += (size_t)NROWS * HIDD;
    float* gbuf   = ws + off; off += (size_t)NROWS * HIDD;
    float* pbuf   = ws + off; off += (size_t)NROWS * POUT;
    Spline* uspl  = (Spline*)(ws + off); off += (64 * sizeof(Spline) + 3) / 4 + 4;
    _Float16* w16 = (_Float16*)(ws + off);

    // per-block f16 weight layout (halves)
    const size_t SZ_WIN = 256 * KINPAD;             // 73728
    const size_t SZ_SQ  = 256 * 256;                // 65536
    const size_t BLK16  = SZ_WIN + 6 * SZ_SQ + SZ_SQ;   // Win + W1/W2/Wc x2 + Wf(pad)
    auto w16_win = [&](int i)        { return w16 + i * BLK16; };
    auto w16_w1  = [&](int i, int j) { return w16 + i * BLK16 + SZ_WIN + j * SZ_SQ; };
    auto w16_w2  = [&](int i, int j) { return w16 + i * BLK16 + SZ_WIN + (2 + j) * SZ_SQ; };
    auto w16_wc  = [&](int i, int j) { return w16 + i * BLK16 + SZ_WIN + (4 + j) * SZ_SQ; };
    auto w16_wf  = [&](int i)        { return w16 + i * BLK16 + SZ_WIN + 6 * SZ_SQ; };

    const dim3 rowGrid(NROWS / 256), rowBlk(256);
    const dim3 gGrid(1, NROWS / 64), gBlk(256);

    init_state<<<rowGrid, rowBlk, 0, stream>>>(x, zbuf, logdet);
    build_uncond<<<1, 64, 0, stream>>>(unc_w, unc_h, unc_d, uspl);

    // one-time weight conversion to transposed f16
    for (int i = 0; i < NBLK; ++i) {
        conv_w16<<<dim3(256 * KINPAD / 256), 256, 0, stream>>>(
            W_in + (size_t)i * KIN * HIDD, w16_win(i), KIN, HIDD, KINPAD);
        for (int j = 0; j < 2; ++j) {
            size_t wo = (size_t)i * 2 + j;
            conv_w16<<<dim3(256), 256, 0, stream>>>(
                W1 + wo * SZ_SQ, w16_w1(i, j), HIDD, HIDD, HIDD);
            conv_w16<<<dim3(256), 256, 0, stream>>>(
                W2 + wo * SZ_SQ, w16_w2(i, j), HIDD, HIDD, HIDD);
            conv_w16<<<dim3(256), 256, 0, stream>>>(
                Wc + wo * SZ_SQ, w16_wc(i, j), HIDD, HIDD, HIDD);
        }
        conv_w16<<<dim3(256), 256, 0, stream>>>(
            Wf + (size_t)i * HIDD * POUT, w16_wf(i), HIDD, POUT, HIDD);
    }

    for (int i = NBLK - 1; i >= 0; --i) {
        prep_block<<<rowGrid, rowBlk, 0, stream>>>(
            zbuf, perms + i * LATD,
            lu_lo + i * 120, lu_up + i * 120, lu_ud + i * LATD, lu_b + i * LATD,
            uspl + i * D_ID, zidb, ztrb, logdet);

        // hcur = concat(zid, ctx) @ W_in[i] + b_in[i]
        gemm_wmma<0, false, true><<<gGrid, gBlk, 0, stream>>>(
            zidb, ctx, w16_win(i), b_in + i * HIDD, nullptr, hcur,
            KIN, KINPAD, HIDD);

        for (int j = 0; j < 2; ++j) {
            size_t wo = (size_t)i * 2 + j;
            // g = ctx @ Wc + bc
            gemm_wmma<0, false, false><<<gGrid, gBlk, 0, stream>>>(
                ctx, nullptr, w16_wc(i, j), bc + wo * HIDD, nullptr, gbuf,
                CTXD, CTXD, HIDD);
            // t1 = relu(hcur) @ W1 + b1
            gemm_wmma<0, true, false><<<gGrid, gBlk, 0, stream>>>(
                hcur, nullptr, w16_w1(i, j), b1 + wo * HIDD, nullptr, t1b,
                HIDD, HIDD, HIDD);
            // hcur += (relu(t1) @ W2 + b2) * sigmoid(g)
            gemm_wmma<1, true, false><<<gGrid, gBlk, 0, stream>>>(
                t1b, nullptr, w16_w2(i, j), b2 + wo * HIDD, gbuf, hcur,
                HIDD, HIDD, HIDD);
        }

        // p = hcur @ Wf[i] + bf[i]
        gemm_wmma<0, false, false><<<gGrid, gBlk, 0, stream>>>(
            hcur, nullptr, w16_wf(i), bf + i * POUT, nullptr, pbuf,
            HIDD, HIDD, POUT);

        finish_block<<<rowGrid, rowBlk, 0, stream>>>(pbuf, zidb, ztrb, zbuf, logdet);
    }

    final_gauss<<<rowGrid, rowBlk, 0, stream>>>(ctx, q0W, q0b, zbuf, logdet, out);
    (void)in_sizes; (void)n_in; (void)out_size; (void)ws_size;
}
